// DeformGNN_20220706030027
// MI455X (gfx1250) — matrix operand order repaired and checked by reading, not compiled, not run on hardware
//
#include <hip/hip_runtime.h>
#include <hip/hip_bf16.h>

// ---------------------------------------------------------------------------
// DeformGNN forward on gfx1250 (MI455X).
// bf16 WMMA GEMMs, double-buffered LDS fed by GLOBAL_LOAD_ASYNC_TO_LDS_B128.
// ---------------------------------------------------------------------------

typedef __attribute__((ext_vector_type(16))) __bf16 v16bf;
typedef __attribute__((ext_vector_type(8)))  float  v8f;

union Frag16 { v16bf v; uint4 q[2]; unsigned short u[16]; };

#define GF_BIAS  1
#define GF_ADD0  2
#define GF_ADD1  4
#define GF_RELU  8
#define GF_BF16  16   // row-major bf16 shadow (next GEMM's A operand)
#define GF_BF16T 32   // transposed bf16 shadow (next GEMM's B operand)

__device__ __forceinline__ unsigned short f32_to_bf16_rne(float f) {
    unsigned int u = __float_as_uint(f);
    u = (u + 0x7FFFu + ((u >> 16) & 1u)) >> 16;
    return (unsigned short)u;
}

// Async global -> LDS copy, 16B per lane, tracked by ASYNCcnt.
__device__ __forceinline__ void async_g2l_b128(unsigned lds_off, const void* gptr) {
    asm volatile("global_load_async_to_lds_b128 %0, %1, off"
                 :: "v"(lds_off), "v"((unsigned long long)(uintptr_t)gptr)
                 : "memory");
}
__device__ __forceinline__ void wait_async0() {
    asm volatile("s_wait_asynccnt 0x0" ::: "memory");
}

// ---------------------------------------------------------------------------
// Row-major pad+convert (adjacency): out[m][r][c] = bf16(in[m][r][c])
// ---------------------------------------------------------------------------
__global__ void pad_convert_kernel(const float* __restrict__ in,
                                   unsigned short* __restrict__ out,
                                   int rin, int rout, int cols, long long total) {
    long long idx = (long long)blockIdx.x * 256 + threadIdx.x;
    if (idx >= total) return;
    long long col = idx % cols;
    long long r   = (idx / cols) % rout;
    long long m   = idx / ((long long)cols * rout);
    float v = (r < rin) ? in[(m * rin + r) * cols + col] : 0.0f;
    out[idx] = f32_to_bf16_rne(v);
}

// ---------------------------------------------------------------------------
// Transposing pad+convert (weights): in [m][rin][cols] -> out [m][cols][rout]
// ---------------------------------------------------------------------------
__global__ void pad_convert_t_kernel(const float* __restrict__ in,
                                     unsigned short* __restrict__ out,
                                     int rin, int rout, int cols, long long total) {
    long long idx = (long long)blockIdx.x * 256 + threadIdx.x;
    if (idx >= total) return;
    long long r = idx % rout;
    long long c = (idx / rout) % cols;
    long long m = idx / ((long long)rout * cols);
    float v = (r < rin) ? in[(m * rin + r) * cols + c] : 0.0f;
    out[idx] = f32_to_bf16_rne(v);
}

// ---------------------------------------------------------------------------
// Bilinear interp + concat + pad -> x0 bf16, emitted row-major AND transposed
// ---------------------------------------------------------------------------
__global__ void interp_kernel(const float* __restrict__ feat,
                              const float* __restrict__ base,
                              unsigned short* __restrict__ x0n,   // [B][N][160]
                              unsigned short* __restrict__ x0t,   // [B][160][N]
                              int Bt, int Np) {
    const int KINc = 160, Cc = 128, Hh = 256, Ww = 256;
    long long idx = (long long)blockIdx.x * 256 + threadIdx.x;
    long long total = (long long)Bt * Np * KINc;
    if (idx >= total) return;
    int c = (int)(idx % KINc);
    long long bn = idx / KINc;
    int n = (int)(bn % Np);
    int b = (int)(bn / Np);
    float px = base[bn * 2 + 0];
    float py = base[bn * 2 + 1];
    float val;
    if (c < Cc) {
        float xs = px * (float)Ww, ys = py * (float)Hh;
        float x0f = floorf(xs), y0f = floorf(ys);
        float wx1 = xs - x0f, wy1 = ys - y0f;
        float wx0 = 1.0f - wx1, wy0 = 1.0f - wy1;
        int X0 = min(max((int)x0f, 0), Ww - 1);
        int X1 = min(max((int)(x0f + 1.0f), 0), Ww - 1);
        int Y0 = min(max((int)y0f, 0), Hh - 1);
        int Y1 = min(max((int)(y0f + 1.0f), 0), Hh - 1);
        const float* fb = feat + ((size_t)b * Cc + c) * (size_t)(Hh * Ww);
        val = wx0 * wy0 * fb[Y0 * Ww + X0] + wx0 * wy1 * fb[Y1 * Ww + X0] +
              wx1 * wy0 * fb[Y0 * Ww + X1] + wx1 * wy1 * fb[Y1 * Ww + X1];
    } else if (c == Cc)     val = px;
    else if (c == Cc + 1)   val = py;
    else                    val = 0.0f;
    unsigned short bv = f32_to_bf16_rne(val);
    x0n[idx] = bv;
    x0t[((size_t)b * KINc + c) * Np + n] = bv;
}

// ---------------------------------------------------------------------------
// bf16 WMMA GEMM:  C[b] = A[b] @ B[b] (+bias)(+add0)(+add1)(relu)
//   A : row-major [M x K] bf16
//   Bt: TRANSPOSED [Ncol x K] bf16 (K contiguous) -> async-copy friendly
// Block tile 256x32, K-step 32, double-buffered LDS, async global->LDS,
// 8 waves x (32x32 strip) = 4 v_wmma_f32_16x16x32_bf16 per wave per K-step.
// Requires M%256==0, Ncol%32==0, K%32==0 (holds for this model).
// ---------------------------------------------------------------------------
__global__ __launch_bounds__(256)
void gemm_bf16_wmma(const unsigned short* __restrict__ A, int lda, unsigned long long sA,
                    const unsigned short* __restrict__ Bt, int ldbt, unsigned long long sB,
                    const float* __restrict__ bias,
                    const float* __restrict__ add0, const float* __restrict__ add1,
                    float* __restrict__ Cf, unsigned short* __restrict__ Cbf,
                    unsigned short* __restrict__ CbT,
                    int M, int K, int Ncol, unsigned long long sC, int flags) {
    // pitch 40 ushorts (80B): keeps 16B alignment, skews LDS banks
    __shared__ unsigned short As[2 * 256 * 40];
    __shared__ unsigned short Bs[2 * 32 * 40];
    const unsigned asBytes = (unsigned)(256 * 40 * 2);
    const unsigned bsBytes = (unsigned)(32 * 40 * 2);
    const unsigned asLds = (unsigned)(uintptr_t)(void*)As;
    const unsigned bsLds = (unsigned)(uintptr_t)(void*)Bs;

    const int tid  = threadIdx.x;
    const int lane = tid & 31;
    const int wave = tid >> 5;
    const int n0 = blockIdx.x * 32;
    const int m0 = blockIdx.y * 256;
    const int bz = blockIdx.z;

    const unsigned short* Ab = A  + (size_t)bz * sA;
    const unsigned short* Bb = Bt + (size_t)bz * sB;

    v8f acc00 = {}, acc01 = {}, acc10 = {}, acc11 = {};

    // stage one 256x32 A tile + 32x32 B tile into LDS buffer `buf` (all async)
    auto stage = [&](int k0, int buf) {
        unsigned aBase = asLds + (unsigned)buf * asBytes;
        unsigned bBase = bsLds + (unsigned)buf * bsBytes;
#pragma unroll
        for (int s = 0; s < 4; ++s) {
            int i = tid + s * 256;           // 1024 16B chunks: 256 rows x 4
            int row = i >> 2, ch = (i & 3) * 8;
            async_g2l_b128(aBase + (unsigned)(row * 40 + ch) * 2,
                           Ab + (size_t)(m0 + row) * lda + k0 + ch);
        }
        if (tid < 128) {                     // 128 16B chunks: 32 rows x 4
            int row = tid >> 2, ch = (tid & 3) * 8;
            async_g2l_b128(bBase + (unsigned)(row * 40 + ch) * 2,
                           Bb + (size_t)(n0 + row) * ldbt + k0 + ch);
        }
    };

    stage(0, 0);
    wait_async0();
    __syncthreads();

    // per-wave fragment addresses (constant across K loop)
    const int r0    = wave * 32;
    const int arow  = r0 + (lane & 15);
    const int khalf = (lane >> 4) * 8;
    const int bcol  = lane & 15;
    const int kh    = (lane >> 4) * 16;

    int cur = 0;
    for (int k0 = 0; k0 < K; k0 += 32) {
        const bool has_next = (k0 + 32) < K;
        if (has_next) stage(k0 + 32, cur ^ 1);

        const unsigned short* Ac = As + cur * (256 * 40);
        const unsigned short* Bc = Bs + cur * (32 * 40);

        Frag16 a0, a1, b0, b1;
        a0.q[0] = *(const uint4*)&Ac[arow * 40 + khalf];
        a0.q[1] = *(const uint4*)&Ac[arow * 40 + 16 + khalf];
        a1.q[0] = *(const uint4*)&Ac[(arow + 16) * 40 + khalf];
        a1.q[1] = *(const uint4*)&Ac[(arow + 16) * 40 + 16 + khalf];
        b0.q[0] = *(const uint4*)&Bc[bcol * 40 + khalf];
        b0.q[1] = *(const uint4*)&Bc[bcol * 40 + 16 + khalf];
        b1.q[0] = *(const uint4*)&Bc[(16 + bcol) * 40 + khalf];
        b1.q[1] = *(const uint4*)&Bc[(16 + bcol) * 40 + 16 + khalf];

        acc00 = __builtin_amdgcn_wmma_f32_16x16x32_bf16(false, a0.v, false, b0.v,
                                                        (short)0, acc00, false, false);
        acc01 = __builtin_amdgcn_wmma_f32_16x16x32_bf16(false, a0.v, false, b1.v,
                                                        (short)0, acc01, false, false);
        acc10 = __builtin_amdgcn_wmma_f32_16x16x32_bf16(false, a1.v, false, b0.v,
                                                        (short)0, acc10, false, false);
        acc11 = __builtin_amdgcn_wmma_f32_16x16x32_bf16(false, a1.v, false, b1.v,
                                                        (short)0, acc11, false, false);

        if (has_next) {
            wait_async0();       // our async writes into buf^1 have landed
            __syncthreads();     // everyone done reading buf `cur`, writes visible
            cur ^= 1;
        }
    }

    // --- epilogue: C/D layout = 8 rows per half-wave, col per lane ---
    const int col   = lane & 15;
    const int rbase = (lane >> 4) * 8;
    float* Cb = Cf + (size_t)bz * sC;
    unsigned short* Cn = (flags & GF_BF16)  ? Cbf + (size_t)bz * sC : nullptr;
    unsigned short* Ct = (flags & GF_BF16T) ? CbT + (size_t)bz * sC : nullptr;
    const float* a0p = (flags & GF_ADD0) ? add0 + (size_t)bz * sC : nullptr;
    const float* a1p = (flags & GF_ADD1) ? add1 + (size_t)bz * sC : nullptr;

#pragma unroll
    for (int tr = 0; tr < 2; ++tr) {
#pragma unroll
        for (int tc = 0; tc < 2; ++tc) {
            const v8f& acc = tr ? (tc ? acc11 : acc10) : (tc ? acc01 : acc00);
            int n = n0 + tc * 16 + col;
            if (n >= Ncol) continue;
            float bv = (flags & GF_BIAS) ? bias[n] : 0.0f;
#pragma unroll
            for (int v = 0; v < 8; ++v) {
                int m = m0 + r0 + tr * 16 + rbase + v;
                if (m >= M) continue;
                float val = acc[v] + bv;
                size_t o = (size_t)m * Ncol + n;
                if (flags & GF_ADD0) val += a0p[o];
                if (flags & GF_ADD1) val += a1p[o];
                if (flags & GF_RELU) val = fmaxf(val, 0.0f);
                Cb[o] = val;
                unsigned short h = f32_to_bf16_rne(val);
                if (flags & GF_BF16)  Cn[o] = h;
                if (flags & GF_BF16T) Ct[(size_t)n * M + m] = h;
            }
        }
    }
}

// ---------------------------------------------------------------------------
// Head: gcn_pred = x @ w_fc + b_fc ; pred = base + gcn_pred*mask
// ---------------------------------------------------------------------------
__global__ void head_kernel(const float* __restrict__ x,
                            const float* __restrict__ w_fc,
                            const float* __restrict__ b_fc,
                            const float* __restrict__ base,
                            const float* __restrict__ pmask,
                            float* __restrict__ out,
                            float* __restrict__ dvec,
                            int Np, int Bt) {
    int idx = blockIdx.x * 256 + threadIdx.x;
    if (idx >= Bt * Np) return;
    const float* xr = x + (size_t)idx * 256;
    float p0 = b_fc[0], p1 = b_fc[1];
    for (int c = 0; c < 256; ++c) {
        float f = xr[c];
        p0 += f * w_fc[c * 2 + 0];
        p1 += f * w_fc[c * 2 + 1];
    }
    float mk = pmask[idx];
    float d0 = p0 * mk, d1 = p1 * mk;
    out[idx * 2 + 0] = base[idx * 2 + 0] + d0;
    out[idx * 2 + 1] = base[idx * 2 + 1] + d1;
    float* gp = out + (size_t)Bt * Np * 2;
    gp[idx * 2 + 0] = p0;
    gp[idx * 2 + 1] = p1;
    dvec[idx * 2 + 0] = d0;
    dvec[idx * 2 + 1] = d1;
}

// ---------------------------------------------------------------------------
// Laplacian energy (deterministic two-stage): (lap2-lap1) == d - adj@d
// ---------------------------------------------------------------------------
__global__ void lap_row_kernel(const float* __restrict__ adj,
                               const float* __restrict__ dvec,
                               float* __restrict__ rowv, int Np) {
    __shared__ float s0[256], s1[256];
    int bi = blockIdx.x;
    int b = bi / Np, i = bi % Np;
    const float* arow = adj + ((size_t)b * Np + i) * Np;
    const float* db = dvec + (size_t)b * Np * 2;
    float a0 = 0.0f, a1 = 0.0f;
    for (int j = threadIdx.x; j < Np; j += 256) {
        float w = arow[j];
        a0 += w * db[j * 2 + 0];
        a1 += w * db[j * 2 + 1];
    }
    s0[threadIdx.x] = a0; s1[threadIdx.x] = a1;
    __syncthreads();
    for (int s = 128; s > 0; s >>= 1) {
        if (threadIdx.x < s) { s0[threadIdx.x] += s0[threadIdx.x + s];
                               s1[threadIdx.x] += s1[threadIdx.x + s]; }
        __syncthreads();
    }
    if (threadIdx.x == 0) {
        float e0 = db[i * 2 + 0] - s0[0];
        float e1 = db[i * 2 + 1] - s1[0];
        rowv[bi] = sqrtf(e0 * e0 + e1 * e1 + 1e-10f);
    }
}

__global__ void lap_reduce_kernel(const float* __restrict__ rowv,
                                  float* __restrict__ energy, int Np) {
    __shared__ float s0[256];
    int b = blockIdx.x;
    float a = 0.0f;
    for (int j = threadIdx.x; j < Np; j += 256) a += rowv[(size_t)b * Np + j];
    s0[threadIdx.x] = a;
    __syncthreads();
    for (int s = 128; s > 0; s >>= 1) {
        if (threadIdx.x < s) s0[threadIdx.x] += s0[threadIdx.x + s];
        __syncthreads();
    }
    if (threadIdx.x == 0) energy[b] = s0[0] / (float)Np;
}

// ---------------------------------------------------------------------------
// Host orchestration
// ---------------------------------------------------------------------------
extern "C" void kernel_launch(void* const* d_in, const int* in_sizes, int n_in,
                              void* d_out, int out_size, void* d_ws, size_t ws_size,
                              hipStream_t stream) {
    (void)in_sizes; (void)n_in; (void)out_size; (void)ws_size;

    const float* features   = (const float*)d_in[0];
    const float* base_point = (const float*)d_in[1];
    const float* adj        = (const float*)d_in[2];
    const float* point_mask = (const float*)d_in[3];
    const float* w_first    = (const float*)d_in[4];
    const float* b_first    = (const float*)d_in[5];
    const float* w_mid      = (const float*)d_in[6];
    const float* b_mid      = (const float*)d_in[7];
    const float* w_last     = (const float*)d_in[8];
    const float* b_last     = (const float*)d_in[9];
    const float* w_fc       = (const float*)d_in[10];
    const float* b_fc       = (const float*)d_in[11];

    const int Bt = 2, Np = 2048, S = 256, KIN = 160, FEAT_IN = 130, MIDN = 6;
    const size_t NN = (size_t)Np * Np;
    const size_t NS = (size_t)Np * S;
    const size_t NK = (size_t)Np * KIN;

    char* ws = (char*)d_ws;
    size_t off = 0;
    auto alloc = [&](size_t bytes) -> void* {
        void* p = ws + off;
        off = (off + bytes + 255) & ~(size_t)255;
        return p;
    };
    unsigned short* adj_bf   = (unsigned short*)alloc((size_t)Bt * NN * 2);
    unsigned short* wfirstT  = (unsigned short*)alloc((size_t)2 * S * KIN * 2);
    unsigned short* wmidT    = (unsigned short*)alloc((size_t)MIDN * 4 * S * S * 2);
    unsigned short* wlastT   = (unsigned short*)alloc((size_t)2 * S * S * 2);
    unsigned short* x0_bf    = (unsigned short*)alloc((size_t)Bt * NK * 2);
    unsigned short* x0T      = (unsigned short*)alloc((size_t)Bt * NK * 2);
    float*          t_f      = (float*)alloc((size_t)Bt * NS * 4);
    unsigned short* t_bf     = (unsigned short*)alloc((size_t)Bt * NS * 2);
    float*          u_f      = (float*)alloc((size_t)Bt * NS * 4);
    float*          xA_f     = (float*)alloc((size_t)Bt * NS * 4);
    unsigned short* xA_bf    = (unsigned short*)alloc((size_t)Bt * NS * 2);
    unsigned short* xA_T     = (unsigned short*)alloc((size_t)Bt * NS * 2);
    float*          xB_f     = (float*)alloc((size_t)Bt * NS * 4);
    unsigned short* xB_bf    = (unsigned short*)alloc((size_t)Bt * NS * 2);
    unsigned short* xB_T     = (unsigned short*)alloc((size_t)Bt * NS * 2);
    float*          h_f      = (float*)alloc((size_t)Bt * NS * 4);
    unsigned short* h_bf     = (unsigned short*)alloc((size_t)Bt * NS * 2);
    unsigned short* h_T      = (unsigned short*)alloc((size_t)Bt * NS * 2);
    float*          dvec     = (float*)alloc((size_t)Bt * Np * 2 * 4);
    float*          rowv     = (float*)alloc((size_t)Bt * Np * 4);

    auto nblk = [](long long t) { return (unsigned)((t + 255) / 256); };

    // --- conversions (adj row-major; weights transposed+padded) ---
    {
        long long t;
        t = (long long)Bt * Np * Np;
        pad_convert_kernel<<<nblk(t), 256, 0, stream>>>(adj, adj_bf, Np, Np, Np, t);
        t = (long long)2 * S * KIN;
        pad_convert_t_kernel<<<nblk(t), 256, 0, stream>>>(w_first, wfirstT, FEAT_IN, KIN, S, t);
        t = (long long)MIDN * 4 * S * S;
        pad_convert_t_kernel<<<nblk(t), 256, 0, stream>>>(w_mid, wmidT, S, S, S, t);
        t = (long long)2 * S * S;
        pad_convert_t_kernel<<<nblk(t), 256, 0, stream>>>(w_last, wlastT, S, S, S, t);
        t = (long long)Bt * Np * KIN;
        interp_kernel<<<nblk(t), 256, 0, stream>>>(features, base_point, x0_bf, x0T, Bt, Np);
    }

    auto gemm = [&](const unsigned short* A, int lda, size_t sA,
                    const unsigned short* Btp, int ldbt, size_t sB,
                    const float* bias, const float* add0, const float* add1,
                    float* Cf, unsigned short* Cn, unsigned short* Ct,
                    int M, int K, int Ncol, int flags) {
        dim3 grid((unsigned)(Ncol / 32), (unsigned)(M / 256), (unsigned)Bt);
        gemm_bf16_wmma<<<grid, 256, 0, stream>>>(
            A, lda, (unsigned long long)sA, Btp, ldbt, (unsigned long long)sB,
            bias, add0, add1, Cf, Cn, Ct, M, K, Ncol,
            (unsigned long long)M * (unsigned long long)Ncol, flags);
    };

    // --- first gconv: x = x0@W1 + b1 + (adj@x0)@W2 + b2 ---
    gemm(adj_bf, Np, NN, x0T, Np, NK, nullptr, nullptr, nullptr,
         t_f, t_bf, nullptr, Np, Np, KIN, GF_BF16);
    gemm(x0_bf, KIN, NK, wfirstT, KIN, 0, b_first, nullptr, nullptr,
         u_f, nullptr, nullptr, Np, KIN, S, GF_BIAS);
    gemm(t_bf, KIN, NK, wfirstT + (size_t)S * KIN, KIN, 0, b_first + S, u_f, nullptr,
         xA_f, xA_bf, xA_T, Np, KIN, S, GF_BIAS | GF_ADD0 | GF_BF16 | GF_BF16T);

    float* curF = xA_f; unsigned short* curB = xA_bf; unsigned short* curT = xA_T;
    float* nxtF = xB_f; unsigned short* nxtB = xB_bf; unsigned short* nxtT = xB_T;

    // --- 6 residual blocks ---
    for (int i = 0; i < MIDN; ++i) {
        const unsigned short* w = wmidT + (size_t)i * 4 * S * S;
        const float* bb = b_mid + (size_t)i * 4 * S;
        // h = relu(gconv1(x))
        gemm(adj_bf, Np, NN, curT, Np, NS, nullptr, nullptr, nullptr,
             t_f, t_bf, nullptr, Np, Np, S, GF_BF16);
        gemm(curB, S, NS, w, S, 0, bb, nullptr, nullptr,
             u_f, nullptr, nullptr, Np, S, S, GF_BIAS);
        gemm(t_bf, S, NS, w + (size_t)S * S, S, 0, bb + S, u_f, nullptr,
             h_f, h_bf, h_T, Np, S, S, GF_BIAS | GF_ADD0 | GF_RELU | GF_BF16 | GF_BF16T);
        // x' = relu(gconv2(h) + x)
        gemm(adj_bf, Np, NN, h_T, Np, NS, nullptr, nullptr, nullptr,
             t_f, t_bf, nullptr, Np, Np, S, GF_BF16);
        gemm(h_bf, S, NS, w + (size_t)2 * S * S, S, 0, bb + 2 * S, nullptr, nullptr,
             u_f, nullptr, nullptr, Np, S, S, GF_BIAS);
        gemm(t_bf, S, NS, w + (size_t)3 * S * S, S, 0, bb + 3 * S, u_f, curF,
             nxtF, nxtB, nxtT, Np, S, S,
             GF_BIAS | GF_ADD0 | GF_ADD1 | GF_RELU | GF_BF16 | GF_BF16T);
        { float* tf = curF; curF = nxtF; nxtF = tf; }
        { unsigned short* tb = curB; curB = nxtB; nxtB = tb; }
        { unsigned short* tt = curT; curT = nxtT; nxtT = tt; }
    }

    // --- last gconv (no relu / residual), f32 output only ---
    gemm(adj_bf, Np, NN, curT, Np, NS, nullptr, nullptr, nullptr,
         t_f, t_bf, nullptr, Np, Np, S, GF_BF16);
    gemm(curB, S, NS, wlastT, S, 0, b_last, nullptr, nullptr,
         u_f, nullptr, nullptr, Np, S, S, GF_BIAS);
    gemm(t_bf, S, NS, wlastT + (size_t)S * S, S, 0, b_last + S, u_f, nullptr,
         nxtF, nullptr, nullptr, Np, S, S, GF_BIAS | GF_ADD0);
    float* xfin = nxtF;

    // --- head + laplacian energy ---
    float* out = (float*)d_out;
    head_kernel<<<nblk((long long)Bt * Np), 256, 0, stream>>>(
        xfin, w_fc, b_fc, base_point, point_mask, out, dvec, Np, Bt);
    lap_row_kernel<<<(unsigned)(Bt * Np), 256, 0, stream>>>(adj, dvec, rowv, Np);
    lap_reduce_kernel<<<(unsigned)Bt, 256, 0, stream>>>(
        rowv, out + (size_t)2 * Bt * Np * 2, Np);
}
